// ESIM_29480655519926
// MI455X (gfx1250) — compile-verified, matrix operand order checked by reading, not run on hardware
//
#include <hip/hip_runtime.h>
#include <hip/hip_bf16.h>

typedef __bf16 bf16;
typedef __attribute__((ext_vector_type(8)))  bf16  v8bf;
typedef __attribute__((ext_vector_type(16))) bf16  v16bf;
typedef __attribute__((ext_vector_type(8)))  float v8f;

#define BB 32
#define TT 128
#define EE 300
#define HH 300
#define HP 304      // H padded to multiple of 16 (scan tiles)
#define KE 320      // E/H padded to multiple of 32 (WMMA K)
#define H4 1200
#define H4P 1216    // 4H padded to multiple of 32
#define NEGV -10000000.0f

static __device__ __forceinline__ bf16 to_bf16(float f) {
  unsigned u = __builtin_bit_cast(unsigned, f);
  unsigned r = (u + 0x7FFFu + ((u >> 16) & 1u)) >> 16;   // RNE truncate
  unsigned short s = (unsigned short)r;
  return __builtin_bit_cast(bf16, s);
}
static __device__ __forceinline__ float sigf(float x) { return 1.0f / (1.0f + __expf(-x)); }

// Load one 16x32 bf16 WMMA fragment (A layout; B uses same layout with rows = N
// since B is stored transposed, i.e. N x K row-major).
// Lane L holds row (L&15); K elements: e in [0,8) -> k0 + (L>=16?8:0) + e,
// e in [8,16) -> +16.  => two contiguous 16-byte chunks per lane.
static __device__ __forceinline__ v16bf load_frag(const bf16* base, int ld, int row0, int k0, int lane) {
  int r  = row0 + (lane & 15);
  int kb = k0 + ((lane >> 4) << 3);
  const bf16* p = base + (long)r * ld + kb;
  v8bf lo = *(const v8bf*)(p);
  v8bf hi = *(const v8bf*)(p + 16);
  return __builtin_shufflevector(lo, hi, 0,1,2,3,4,5,6,7,8,9,10,11,12,13,14,15);
}
static __device__ __forceinline__ v8f wmma_bf16(v16bf a, v16bf b, v8f c) {
  return __builtin_amdgcn_wmma_f32_16x16x32_bf16(false, a, false, b, (short)0, c, false, false);
}

// ---------------------------------------------------------------------------
// Generic batched WMMA GEMM: C[b] = A[b] (MxK) * B[b]^T (NxK) + bias, act.
// Each wave computes a 32x32 macro-tile (2x2 WMMA fragments), grid-stride.
// act: 0=none, 1=relu, 2=tanh. Stores val for n<N, zero for n in [N,padN).
// ---------------------------------------------------------------------------
__global__ void k_gemm(const bf16* __restrict__ A, int lda, long sA,
                       const bf16* __restrict__ B, int ldb, long sB,
                       float* __restrict__ Cf, bf16* __restrict__ Cb,
                       int ldc, long sC,
                       int M, int N, int Kp, int batch,
                       const float* __restrict__ bias, int act, int padN)
{
  const int lane = threadIdx.x & 31;
  const int wv   = (blockIdx.x * (blockDim.x >> 5)) + (threadIdx.x >> 5);
  const int nwv  = gridDim.x * (blockDim.x >> 5);
  const int mt = (M + 31) >> 5;
  const int nt = (N + 31) >> 5;
  const long tiles = (long)mt * nt * batch;
  for (long tile = wv; tile < tiles; tile += nwv) {
    int bidx = (int)(tile / ((long)mt * nt));
    int rem  = (int)(tile % ((long)mt * nt));
    int tm = rem / nt, tn = rem % nt;
    const bf16* Ab = A + (long)bidx * sA;
    const bf16* Bb = B + (long)bidx * sB;
    v8f acc00 = {}, acc01 = {}, acc10 = {}, acc11 = {};
    for (int k0 = 0; k0 < Kp; k0 += 32) {
      v16bf a0 = load_frag(Ab, lda, tm * 32,      k0, lane);
      v16bf a1 = load_frag(Ab, lda, tm * 32 + 16, k0, lane);
      v16bf b0 = load_frag(Bb, ldb, tn * 32,      k0, lane);
      v16bf b1 = load_frag(Bb, ldb, tn * 32 + 16, k0, lane);
      acc00 = wmma_bf16(a0, b0, acc00);
      acc01 = wmma_bf16(a0, b1, acc01);
      acc10 = wmma_bf16(a1, b0, acc10);
      acc11 = wmma_bf16(a1, b1, acc11);
    }
    const int half = (lane >> 4) << 3;
    for (int ni = 0; ni < 2; ++ni) {
      int n = tn * 32 + ni * 16 + (lane & 15);
      float bv = (bias && n < N) ? bias[n] : 0.0f;
      for (int mi = 0; mi < 2; ++mi) {
        v8f accv = (ni == 0) ? (mi == 0 ? acc00 : acc10) : (mi == 0 ? acc01 : acc11);
        for (int r2 = 0; r2 < 8; ++r2) {
          int m = tm * 32 + mi * 16 + r2 + half;
          if (m >= M) continue;
          long idx = (long)bidx * sC + (long)m * ldc + n;
          if (n < N) {
            float v = accv[r2] + bv;
            if (act == 1) v = v > 0.0f ? v : 0.0f;
            else if (act == 2) v = tanhf(v);
            if (Cf) Cf[idx] = v;
            if (Cb) Cb[idx] = to_bf16(v);
          } else if (n < padN) {
            if (Cf) Cf[idx] = 0.0f;
            if (Cb) Cb[idx] = to_bf16(0.0f);
          }
        }
      }
    }
  }
}

// Embedding gather -> bf16 padded (B*T x KE)
__global__ void k_embed(const int* __restrict__ tok, const float* __restrict__ emb,
                        bf16* __restrict__ out, int n)
{
  int i = blockIdx.x * blockDim.x + threadIdx.x;
  if (i >= n) return;
  int m = i / KE, e = i % KE;
  float v = 0.0f;
  if (e < EE) v = emb[(long)tok[m] * EE + e];
  out[i] = to_bf16(v);
}

// f32 -> bf16 with zero padding (rows_out x ldout)
__global__ void k_convert(const float* __restrict__ in, int ldin, int rows_in, int cols_in,
                          bf16* __restrict__ out, int ldout, int rows_out)
{
  int i = blockIdx.x * blockDim.x + threadIdx.x;
  int tot = rows_out * ldout;
  if (i >= tot) return;
  int r = i / ldout, c = i % ldout;
  float v = (r < rows_in && c < cols_in) ? in[(long)r * ldin + c] : 0.0f;
  out[i] = to_bf16(v);
}

// (B,T,ldin) f32 -> (B,Hout,T) bf16 transposed, zero padding in h
__global__ void k_tconvert(const float* __restrict__ in, int ldin, int Hin,
                           bf16* __restrict__ out, int Hout)
{
  int i = blockIdx.x * blockDim.x + threadIdx.x;
  int tot = BB * Hout * TT;
  if (i >= tot) return;
  int t  = i % TT;
  int hh = (i / TT) % Hout;
  int b  = i / (TT * Hout);
  float v = (hh < Hin) ? in[((long)b * TT + t) * ldin + hh] : 0.0f;
  out[i] = to_bf16(v);
}

// Precompute gates from XG (B*T x H4P): h_j is identically zero in the
// reference scan, so i/o/u depend only on xg + bias.
// iu = sigmoid(xg0+bh0)*tanh(xg3+bh3), og = sigmoid(xg2+bh2), fx = xg1+bh1.
__global__ void k_gates(const float* __restrict__ XG, const float* __restrict__ bh,
                        float* __restrict__ iu, float* __restrict__ og, float* __restrict__ fx)
{
  int i = blockIdx.x * blockDim.x + threadIdx.x;
  int tot = BB * TT * HH;
  if (i >= tot) return;
  int m = i / HH, h = i % HH;
  const float* row = XG + (long)m * H4P;
  float iv = sigf(row[h] + bh[h]);
  float fv = row[HH + h] + bh[HH + h];
  float ov = sigf(row[2 * HH + h] + bh[2 * HH + h]);
  float uv = tanhf(row[3 * HH + h] + bh[3 * HH + h]);
  long o = (long)m * HP + h;
  iu[o] = iv * uv;
  og[o] = ov;
  fx[o] = fv;
}

// ---------------------------------------------------------------------------
// Persistent tree-LSTM scan. grid.x = 2 (side 0 = premise/lgraph, side 1 =
// hypothesis/rgraph), 1024 threads = 32 waves. Per step j:
//   phase 1: wave b computes c_j/h_j using sparse graph-gated sum over t<j
//            (f = sigmoid(fx_j + Hu_t)), writes h_j (bf16) into LDS
//   phase 2: Hu[:,j] = h_j @ uh1^T via 38 WMMA 16x16x32 tiles (M=32,N=304,K=320)
// ---------------------------------------------------------------------------
__global__ void __launch_bounds__(1024) k_scan(
    const float* __restrict__ iu, const float* __restrict__ og, const float* __restrict__ fxg,
    const float* __restrict__ lgraph, const float* __restrict__ rgraph,
    const bf16* __restrict__ uh1,
    float* __restrict__ hout, float* __restrict__ cbuf, float* __restrict__ Hu)
{
  __shared__ bf16 lds_h[BB * KE];
  const long ss = (long)BB * TT * HP * blockIdx.x;
  const float* graph = blockIdx.x ? rgraph : lgraph;
  iu += ss; og += ss; fxg += ss; hout += ss; cbuf += ss; Hu += ss;

  const int tid  = threadIdx.x;
  const int lane = tid & 31;
  const int wv   = tid >> 5;

  // zero the K-padding columns (300..319) of the LDS h-tile once
  for (int i = tid; i < BB * (KE - HH); i += blockDim.x) {
    int row = i / (KE - HH), col = HH + i % (KE - HH);
    lds_h[row * KE + col] = to_bf16(0.0f);
  }
  __syncthreads();

  for (int j = 0; j < TT; ++j) {
    { // phase 1: wave wv owns batch row b = wv
      const int b = wv;
      float acc[10], fxv[10];
      const long rowj = ((long)b * TT + j) * HP;
      for (int r = 0; r < 10; ++r) {
        int h = lane + 32 * r;
        if (h < HH) { acc[r] = iu[rowj + h]; fxv[r] = fxg[rowj + h]; }
        else        { acc[r] = 0.0f;         fxv[r] = 0.0f; }
      }
      const float* grow = graph + ((long)b * TT + j) * TT;
      for (int t = 0; t < j; ++t) {
        float g = grow[t];                       // ~1% nonzero
        if (g != 0.0f) {
          const float* hu = Hu   + ((long)b * TT + t) * HP;
          const float* cr = cbuf + ((long)b * TT + t) * HP;
          for (int r = 0; r < 10; ++r) {
            int h = lane + 32 * r;
            if (h < HH) acc[r] += g * sigf(fxv[r] + hu[h]) * cr[h];
          }
        }
      }
      for (int r = 0; r < 10; ++r) {
        int h = lane + 32 * r;
        if (h < HH) {
          float cj = acc[r];
          cbuf[rowj + h] = cj;
          float hj = og[rowj + h] * tanhf(cj);
          hout[rowj + h] = hj;
          lds_h[b * KE + h] = to_bf16(hj);
        }
      }
    }
    __syncthreads();
    // phase 2: Hu[:, j] = h_j @ uh1^T ; 2 m-tiles x 19 n-tiles over 32 waves
    for (int tidx = wv; tidx < 2 * (HP / 16); tidx += 32) {
      int mi = tidx & 1;
      int ni = tidx >> 1;
      v8f acc = {};
      for (int k0 = 0; k0 < KE; k0 += 32) {
        v16bf a = load_frag(lds_h, KE, mi * 16, k0, lane);
        v16bf b = load_frag(uh1,   KE, ni * 16, k0, lane);
        acc = wmma_bf16(a, b, acc);
      }
      int n = ni * 16 + (lane & 15);
      int half = (lane >> 4) << 3;
      for (int r = 0; r < 8; ++r) {
        int brow = mi * 16 + r + half;
        Hu[((long)brow * TT + j) * HP + n] = acc[r];
      }
    }
    __syncthreads();
  }
}

// Masked softmax over 128 keys; transposed=1 reads sim^T. Output bf16 probs.
__global__ void k_softmax(const float* __restrict__ sim, const int* __restrict__ lens,
                          bf16* __restrict__ probs, int transposed)
{
  __shared__ float red[TT];
  int b   = blockIdx.x >> 7;
  int row = blockIdx.x & 127;
  int q   = threadIdx.x;
  const float* sb = sim + (long)b * TT * TT;
  float v = transposed ? sb[(long)q * TT + row] : sb[(long)row * TT + q];
  if (q >= lens[b]) v = NEGV;
  red[q] = v; __syncthreads();
  for (int s = 64; s > 0; s >>= 1) { if (q < s) red[q] = fmaxf(red[q], red[q + s]); __syncthreads(); }
  float mx = red[0]; __syncthreads();
  float e = __expf(v - mx);
  red[q] = e; __syncthreads();
  for (int s = 64; s > 0; s >>= 1) { if (q < s) red[q] += red[q + s]; __syncthreads(); }
  float sum = red[0];
  probs[((long)b * TT + row) * TT + q] = to_bf16(e / sum);
}

// enh = [enc, att, enc-att, enc*att] -> bf16 (B*T x H4P), zero pad
__global__ void k_concat(const float* __restrict__ enc, const float* __restrict__ att,
                         bf16* __restrict__ out)
{
  int i = blockIdx.x * blockDim.x + threadIdx.x;
  int tot = BB * TT * H4P;
  if (i >= tot) return;
  int m = i / H4P, c = i % H4P;
  float v = 0.0f;
  if (c < H4) {
    int g = c / HH, h = c % HH;
    float e = enc[(long)m * HP + h];
    float a = att[(long)m * HP + h];
    v = (g == 0) ? e : (g == 1) ? a : (g == 2) ? (e - a) : (e * a);
  }
  out[i] = to_bf16(v);
}

// pooled v = [a_avg, a_max, b_avg, b_max] -> bf16 (32 x H4P), zero pad
__global__ void k_pool(const float* __restrict__ va, const float* __restrict__ vb,
                       const int* __restrict__ plen, const int* __restrict__ hlen,
                       bf16* __restrict__ out)
{
  int b = blockIdx.x;
  int h = threadIdx.x;
  if (h < H4P - H4) out[(long)b * H4P + H4 + h] = to_bf16(0.0f);
  if (h >= HH) return;
  int lp = plen[b], lh = hlen[b];
  float sa = 0.0f, ma = NEGV, sb = 0.0f, mb = NEGV;
  for (int t = 0; t < TT; ++t) {
    float xa = va[((long)b * TT + t) * HP + h];
    float xb = vb[((long)b * TT + t) * HP + h];
    if (t < lp) { sa += xa; ma = fmaxf(ma, xa); }
    if (t < lh) { sb += xb; mb = fmaxf(mb, xb); }
  }
  out[(long)b * H4P + h]          = to_bf16(sa / (float)lp);
  out[(long)b * H4P + HH + h]     = to_bf16(ma);
  out[(long)b * H4P + 2 * HH + h] = to_bf16(sb / (float)lh);
  out[(long)b * H4P + 3 * HH + h] = to_bf16(mb);
}

// logits = hid @ cls_w2^T + cls_b2  (32x3, tiny -> VALU)
__global__ void k_classify(const float* __restrict__ hid, const float* __restrict__ w2,
                           const float* __restrict__ b2, float* __restrict__ out)
{
  int i = threadIdx.x;
  if (i >= BB * 3) return;
  int b = i / 3, c = i % 3;
  float s = b2[c];
  for (int h = 0; h < HH; ++h) s += hid[(long)b * KE + h] * w2[c * HH + h];
  out[i] = s;
}

extern "C" void kernel_launch(void* const* d_in, const int* in_sizes, int n_in,
                              void* d_out, int out_size, void* d_ws, size_t ws_size,
                              hipStream_t stream)
{
  (void)in_sizes; (void)n_in; (void)out_size; (void)ws_size;
  const int*   premises = (const int*)  d_in[0];
  const int*   plen     = (const int*)  d_in[1];
  const int*   hypos    = (const int*)  d_in[2];
  const int*   hlen     = (const int*)  d_in[3];
  const float* lgraph   = (const float*)d_in[4];
  const float* rgraph   = (const float*)d_in[5];
  const float* emb      = (const float*)d_in[6];
  const float* enc_wx   = (const float*)d_in[7];
  const float* enc_bx   = (const float*)d_in[8];
  const float* enc_uh   = (const float*)d_in[9];
  const float* enc_bh   = (const float*)d_in[10];
  const float* comp_wx  = (const float*)d_in[11];
  const float* comp_bx  = (const float*)d_in[12];
  const float* comp_uh  = (const float*)d_in[13];
  const float* comp_bh  = (const float*)d_in[14];
  const float* proj_w   = (const float*)d_in[15];
  const float* proj_b   = (const float*)d_in[16];
  const float* cls_w1   = (const float*)d_in[17];
  const float* cls_b1   = (const float*)d_in[18];
  const float* cls_w2   = (const float*)d_in[19];
  const float* cls_b2   = (const float*)d_in[20];

  char* base = (char*)d_ws;
  size_t off = 0;
  auto alloc = [&](size_t bytes) -> void* {
    void* p = base + off;
    off = (off + bytes + 255) & ~(size_t)255;
    return p;
  };
  const long SS  = (long)BB * TT * HP;      // per-side f32 stride
  const long MT  = (long)BB * TT;           // 4096 rows

  bf16* wxe    = (bf16*)alloc((size_t)H4P * KE * 2);
  bf16* wxc    = (bf16*)alloc((size_t)H4P * KE * 2);
  bf16* uh1e   = (bf16*)alloc((size_t)HP * KE * 2);
  bf16* uh1c   = (bf16*)alloc((size_t)HP * KE * 2);
  bf16* projwB = (bf16*)alloc((size_t)KE * H4P * 2);
  bf16* clsw1B = (bf16*)alloc((size_t)KE * H4P * 2);
  bf16* embp   = (bf16*)alloc((size_t)MT * KE * 2);
  bf16* embh   = (bf16*)alloc((size_t)MT * KE * 2);
  float* XG    = (float*)alloc((size_t)MT * H4P * 4);
  float* iu    = (float*)alloc((size_t)2 * SS * 4);
  float* og    = (float*)alloc((size_t)2 * SS * 4);
  float* fx    = (float*)alloc((size_t)2 * SS * 4);
  float* henc  = (float*)alloc((size_t)2 * SS * 4);
  float* vbuf  = (float*)alloc((size_t)2 * SS * 4);
  float* cscr  = (float*)alloc((size_t)2 * SS * 4);
  float* huscr = (float*)alloc((size_t)2 * SS * 4);
  bf16* encpB  = (bf16*)alloc((size_t)MT * KE * 2);
  bf16* enchB  = (bf16*)alloc((size_t)MT * KE * 2);
  bf16* encpT  = (bf16*)alloc((size_t)BB * KE * TT * 2);
  bf16* enchT  = (bf16*)alloc((size_t)BB * KE * TT * 2);
  float* sim   = (float*)alloc((size_t)BB * TT * TT * 4);
  bf16* probsP = (bf16*)alloc((size_t)BB * TT * TT * 2);
  bf16* probsH = (bf16*)alloc((size_t)BB * TT * TT * 2);
  float* attP  = (float*)alloc((size_t)SS * 4);
  float* attH  = (float*)alloc((size_t)SS * 4);
  bf16* enhP   = (bf16*)alloc((size_t)MT * H4P * 2);
  bf16* enhH   = (bf16*)alloc((size_t)MT * H4P * 2);
  bf16* projPb = (bf16*)alloc((size_t)MT * KE * 2);
  bf16* projHb = (bf16*)alloc((size_t)MT * KE * 2);
  bf16* vpool  = (bf16*)alloc((size_t)BB * H4P * 2);
  float* hid   = (float*)alloc((size_t)BB * KE * 4);

  const int TPB = 256;
  auto cdiv = [](long a, long b) { return (int)((a + b - 1) / b); };
  auto G = [&](const bf16* A, int lda, long sA, const bf16* B, int ldb, long sB,
               float* Cf, bf16* Cb, int ldc, long sC, int M, int N, int Kp, int batch,
               const float* bias, int act, int padN) {
    long tiles = (long)((M + 31) / 32) * ((N + 31) / 32) * batch;
    int blocks = (int)((tiles + 7) / 8);
    if (blocks > 2048) blocks = 2048;
    if (blocks < 1) blocks = 1;
    k_gemm<<<blocks, 256, 0, stream>>>(A, lda, sA, B, ldb, sB, Cf, Cb, ldc, sC,
                                       M, N, Kp, batch, bias, act, padN);
  };

  // --- weight conversion / padding (bf16) ---
  k_convert<<<cdiv((long)H4P * KE, TPB), TPB, 0, stream>>>(enc_wx, EE, H4, EE, wxe, KE, H4P);
  k_convert<<<cdiv((long)H4P * KE, TPB), TPB, 0, stream>>>(comp_wx, HH, H4, HH, wxc, KE, H4P);
  k_convert<<<cdiv((long)HP * KE, TPB), TPB, 0, stream>>>(enc_uh + HH * HH, HH, HH, HH, uh1e, KE, HP);
  k_convert<<<cdiv((long)HP * KE, TPB), TPB, 0, stream>>>(comp_uh + HH * HH, HH, HH, HH, uh1c, KE, HP);
  k_convert<<<cdiv((long)KE * H4P, TPB), TPB, 0, stream>>>(proj_w, H4, HH, H4, projwB, H4P, KE);
  k_convert<<<cdiv((long)KE * H4P, TPB), TPB, 0, stream>>>(cls_w1, H4, HH, H4, clsw1B, H4P, KE);

  // --- embeddings ---
  k_embed<<<cdiv(MT * KE, TPB), TPB, 0, stream>>>(premises, emb, embp, (int)(MT * KE));
  k_embed<<<cdiv(MT * KE, TPB), TPB, 0, stream>>>(hypos, emb, embh, (int)(MT * KE));

  // --- encoder tree-LSTM: xg GEMM + gate precompute + scan ---
  G(embp, KE, 0, wxe, KE, 0, XG, nullptr, H4P, 0, (int)MT, H4, KE, 1, enc_bx, 0, H4);
  k_gates<<<cdiv(MT * HH, TPB), TPB, 0, stream>>>(XG, enc_bh, iu, og, fx);
  G(embh, KE, 0, wxe, KE, 0, XG, nullptr, H4P, 0, (int)MT, H4, KE, 1, enc_bx, 0, H4);
  k_gates<<<cdiv(MT * HH, TPB), TPB, 0, stream>>>(XG, enc_bh, iu + SS, og + SS, fx + SS);
  k_scan<<<2, 1024, 0, stream>>>(iu, og, fx, lgraph, rgraph, uh1e, henc, cscr, huscr);

  // --- attention ---
  k_convert<<<cdiv(MT * KE, TPB), TPB, 0, stream>>>(henc, HP, (int)MT, HH, encpB, KE, (int)MT);
  k_convert<<<cdiv(MT * KE, TPB), TPB, 0, stream>>>(henc + SS, HP, (int)MT, HH, enchB, KE, (int)MT);
  k_tconvert<<<cdiv((long)BB * KE * TT, TPB), TPB, 0, stream>>>(henc, HP, HH, encpT, KE);
  k_tconvert<<<cdiv((long)BB * KE * TT, TPB), TPB, 0, stream>>>(henc + SS, HP, HH, enchT, KE);
  G(encpB, KE, (long)TT * KE, enchB, KE, (long)TT * KE, sim, nullptr, TT, (long)TT * TT,
    TT, TT, KE, BB, nullptr, 0, TT);
  k_softmax<<<BB * TT, TT, 0, stream>>>(sim, hlen, probsP, 0);
  k_softmax<<<BB * TT, TT, 0, stream>>>(sim, plen, probsH, 1);
  // batched attention GEMMs: per-batch C stride = TT*HP
  G(probsP, TT, (long)TT * TT, enchT, TT, (long)KE * TT, attP, nullptr, HP, (long)TT * HP,
    TT, HH, TT, BB, nullptr, 0, HH);
  G(probsH, TT, (long)TT * TT, encpT, TT, (long)KE * TT, attH, nullptr, HP, (long)TT * HP,
    TT, HH, TT, BB, nullptr, 0, HH);

  // --- enh concat + projection (ReLU, bf16 out with zeroed K-padding) ---
  k_concat<<<cdiv(MT * H4P, TPB), TPB, 0, stream>>>(henc, attP, enhP);
  k_concat<<<cdiv(MT * H4P, TPB), TPB, 0, stream>>>(henc + SS, attH, enhH);
  G(enhP, H4P, 0, projwB, H4P, 0, nullptr, projPb, KE, 0, (int)MT, HH, H4P, 1, proj_b, 1, KE);
  G(enhH, H4P, 0, projwB, H4P, 0, nullptr, projHb, KE, 0, (int)MT, HH, H4P, 1, proj_b, 1, KE);

  // --- composition tree-LSTM ---
  G(projPb, KE, 0, wxc, KE, 0, XG, nullptr, H4P, 0, (int)MT, H4, KE, 1, comp_bx, 0, H4);
  k_gates<<<cdiv(MT * HH, TPB), TPB, 0, stream>>>(XG, comp_bh, iu, og, fx);
  G(projHb, KE, 0, wxc, KE, 0, XG, nullptr, H4P, 0, (int)MT, H4, KE, 1, comp_bx, 0, H4);
  k_gates<<<cdiv(MT * HH, TPB), TPB, 0, stream>>>(XG, comp_bh, iu + SS, og + SS, fx + SS);
  k_scan<<<2, 1024, 0, stream>>>(iu, og, fx, lgraph, rgraph, uh1c, vbuf, cscr, huscr);

  // --- pooling + classifier ---
  k_pool<<<BB, KE, 0, stream>>>(vbuf, vbuf + SS, plen, hlen, vpool);
  G(vpool, H4P, 0, clsw1B, H4P, 0, hid, nullptr, KE, 0, BB, HH, H4P, 1, cls_b1, 2, HH);
  k_classify<<<1, 128, 0, stream>>>(hid, cls_w2, cls_b2, (float*)d_out);
}
